// TitanDualStateLayer_70205535420775
// MI455X (gfx1250) — matrix operand order repaired and checked by reading, not compiled, not run on hardware
//
#include <hip/hip_runtime.h>
#include <hip/hip_bf16.h>

// ---------------------------------------------------------------------------
// TitanDualStateLayer for MI455X (gfx1250): bf16 WMMA GEMMs + WMMA chunk scan
// B=4, L=4096, D=1024, H=16, DH=64, CHUNK=128, nC=32
// All WMMA fragment traffic uses 16-byte LDS ops (b128) via fragment-major
// ("swizzled") B-operand layouts; GEMM double-buffers LDS tiles.
// ---------------------------------------------------------------------------

#define Bsz 4
#define Lseq 4096
#define Dmod 1024
#define Hh 16
#define DH 64
#define CHK 128
#define NC 32
#define Mrows (Bsz * Lseq)          // 16384

typedef __attribute__((ext_vector_type(16))) __bf16 v16bf;
typedef __attribute__((ext_vector_type(8)))  __bf16 v8bf;
typedef __attribute__((ext_vector_type(8)))  float  v8f;
typedef __attribute__((ext_vector_type(4)))  unsigned int v4u;

// ------------------------- WMMA fragment helpers ---------------------------
// A fragment: 16x32 bf16 (MxK), row-major source with leading dim `ld`.
// Per lane the 16 values are one contiguous 16-element K run
// (K = khalf..khalf+15)  ->  2x ds_load_b128.
__device__ __forceinline__ v16bf load_frag_a(const __bf16* base, int ld) {
  const int lane  = threadIdx.x & 31;
  const int m     = lane & 15;
  const int khalf = (lane >> 4) << 4;   // 0 or 16
  const v8bf lo = *(const v8bf*)(base + m * ld + khalf);
  const v8bf hi = *(const v8bf*)(base + m * ld + khalf + 8);
  return __builtin_shufflevector(lo, hi, 0, 1, 2, 3, 4, 5, 6, 7,
                                 8, 9, 10, 11, 12, 13, 14, 15);
}

// B fragment from fragment-major storage: a 32x16 (KxN) tile is stored as
// [lane(32)][j(16)] contiguous bf16, so each lane reads 32 contiguous bytes.
// Lane semantics: n = lane&15, k0 = (lane>>4)*16, value j -> element (k0+j, n)
__device__ __forceinline__ v16bf load_frag_b_sw(const __bf16* tile) {
  const int lane = threadIdx.x & 31;
  const v8bf lo = *(const v8bf*)(tile + lane * 16);
  const v8bf hi = *(const v8bf*)(tile + lane * 16 + 8);
  return __builtin_shufflevector(lo, hi, 0, 1, 2, 3, 4, 5, 6, 7,
                                 8, 9, 10, 11, 12, 13, 14, 15);
}

// C/D 16x16 f32: VGPR r -> M=r (lanes 0-15) or M=8+r (lanes 16-31), N=lane%16
__device__ __forceinline__ void store_frag_c(float* base, int ld, v8f c) {
  const int lane = threadIdx.x & 31;
  const int n    = lane & 15;
  const int mo   = (lane >> 4) << 3;
#pragma unroll
  for (int r = 0; r < 8; ++r) base[(size_t)(mo + r) * ld + n] = c[r];
}

__device__ __forceinline__ void store_frag_c_bias(float* base, int ld,
                                                  const float* bcol, v8f c) {
  const int lane = threadIdx.x & 31;
  const int n    = lane & 15;
  const int mo   = (lane >> 4) << 3;
  const float bv = bcol[n];
#pragma unroll
  for (int r = 0; r < 8; ++r) base[(size_t)(mo + r) * ld + n] = c[r] + bv;
}

// Store a 16x16 D tile (tile row dm of a 64-wide K dim, col tile en) into a
// fragment-major B buffer of 32x16 tiles: one ds_store_b128 per lane.
__device__ __forceinline__ void store_frag_sS(__bf16* sS, int dm, int en,
                                              v8f c) {
  const int lane   = threadIdx.x & 31;
  const int n      = lane & 15;
  const int mo     = (lane >> 4) << 3;           // 0 or 8 -> j base
  const int ks     = dm >> 1;                    // 32-row K block
  const int lane_b = n + ((dm & 1) << 4);
  v8bf o;
#pragma unroll
  for (int r = 0; r < 8; ++r) o[r] = (__bf16)c[r];
  *(v8bf*)(&sS[((ks * 4 + en) * 32 + lane_b) * 16 + mo]) = o;
}

__device__ __forceinline__ v8f wmma_bf16(v16bf a, v16bf b, v8f c) {
  return __builtin_amdgcn_wmma_f32_16x16x32_bf16(false, a, false, b,
                                                 (short)0, c, false, false);
}

__device__ __forceinline__ float sigmoidf_(float x) {
  return 1.0f / (1.0f + __expf(-x));
}

// ------------- weight prep: fp32 [N,K] -> bf16 fragment-major --------------
// Layout: [kb(K/32)][ntile(N/16)][lane(32)][j(16)], lane = (n&15)|((k>>4&1)<<4),
// j = k&15. B-tile staging becomes a contiguous copy; frag loads are b128.
__global__ __launch_bounds__(256) void k_prep_w(const float* __restrict__ W,
                                                __bf16* __restrict__ WTsw) {
  const size_t idx = (size_t)blockIdx.x * 256 + threadIdx.x; // < 1M
  const int j     = idx & 15;
  const int lane  = (idx >> 4) & 31;
  const int ntile = (idx >> 9) & 63;
  const int kb    = (int)(idx >> 15);
  const int n = ntile * 16 + (lane & 15);
  const int k = kb * 32 + ((lane >> 4) << 4) + j;
  WTsw[idx] = (__bf16)W[(size_t)n * Dmod + k];
}

// -------------------- depthwise conv (K=4, pad 2, [:L]) + SiLU -------------
__global__ __launch_bounds__(256) void k_conv_silu(const float* __restrict__ x,
                                                   const float* __restrict__ cw,
                                                   const float* __restrict__ cb,
                                                   __bf16* __restrict__ xc) {
  const size_t idx = (size_t)blockIdx.x * 256 + threadIdx.x; // < B*L*D
  const int d = idx & (Dmod - 1);
  const size_t r = idx >> 10;
  const int l = r & (Lseq - 1);
  const int b = (int)(r >> 12);
  float acc = cb[d];
#pragma unroll
  for (int t = 0; t < 4; ++t) {
    const int li = l - 2 + t;
    if (li >= 0 && li < Lseq)
      acc += x[((size_t)b * Lseq + li) * Dmod + d] * cw[d * 4 + t];
  }
  xc[idx] = (__bf16)(acc * sigmoidf_(acc));
}

// ------------------------- bf16 WMMA GEMM + bias ---------------------------
// C[M,N] = A[M,K](bf16 row-major) @ Bsw[K,N](bf16 fragment-major) + bias[N].
// 128x128 block tile, 8 waves (wave32), 2x4 accumulators each; LDS ping-pong.
__device__ __forceinline__ void gemm_stage(const __bf16* __restrict__ A,
                                           const __bf16* __restrict__ bsrc,
                                           __bf16* sA, __bf16* sB,
                                           int m0, int K, int kk, int t) {
#pragma unroll
  for (int i = t; i < 512; i += 256) {
    const int row = i >> 2, c8 = i & 3;
    *(v4u*)(sA + i * 8) =
        *(const v4u*)(A + (size_t)(m0 + row) * K + kk + c8 * 8);
    *(v4u*)(sB + i * 8) = *(const v4u*)(bsrc + i * 8);
  }
}

__global__ __launch_bounds__(256) void k_gemm_bf16(const __bf16* __restrict__ A,
                                                   const __bf16* __restrict__ Bsw,
                                                   const float* __restrict__ bias,
                                                   float* __restrict__ C,
                                                   int K) {
  const int N  = Dmod;
  const int t  = threadIdx.x;
  const int w  = t >> 5;
  const int wm = w & 3;        // 4 row groups of 32
  const int wn = w >> 2;       // 2 col groups of 64
  const int m0 = blockIdx.y * 128;
  const int n0 = blockIdx.x * 128;
  // fragment-major weight block for this column tile: 8 ntiles * 512 elems
  const __bf16* Bblk = Bsw + (size_t)(n0 >> 4) * 512;

  __shared__ __bf16 sA[2][128 * 32];   // 2 x 8 KB
  __shared__ __bf16 sB[2][8 * 512];    // 2 x 8 KB (8 fragment-major tiles)

  v8f acc[2][4];
#pragma unroll
  for (int im = 0; im < 2; ++im)
#pragma unroll
    for (int in = 0; in < 4; ++in)
      acc[im][in] = (v8f){0.f, 0.f, 0.f, 0.f, 0.f, 0.f, 0.f, 0.f};

  const int nk = K >> 5;
  gemm_stage(A, Bblk, sA[0], sB[0], m0, K, 0, t);

  for (int kt = 0; kt < nk; ++kt) {
    __syncthreads();
    const int cur = kt & 1;
    if (kt + 1 < nk) {
      __builtin_prefetch(A + (size_t)(m0 + (t >> 1)) * K + (kt + 1) * 32, 0, 1);
      gemm_stage(A, Bblk + (size_t)(kt + 1) * (64 * 512),
                 sA[cur ^ 1], sB[cur ^ 1], m0, K, (kt + 1) * 32, t);
    }
    const v16bf af0 = load_frag_a(&sA[cur][(wm * 32) * 32], 32);
    const v16bf af1 = load_frag_a(&sA[cur][(wm * 32 + 16) * 32], 32);
#pragma unroll
    for (int in = 0; in < 4; ++in) {
      const v16bf bf = load_frag_b_sw(&sB[cur][(wn * 4 + in) * 512]);
      acc[0][in] = wmma_bf16(af0, bf, acc[0][in]);
      acc[1][in] = wmma_bf16(af1, bf, acc[1][in]);
    }
  }

#pragma unroll
  for (int im = 0; im < 2; ++im)
#pragma unroll
    for (int in = 0; in < 4; ++in) {
      const int gm = m0 + wm * 32 + im * 16;
      const int gn = n0 + wn * 64 + in * 16;
      store_frag_c_bias(C + (size_t)gm * N + gn, N, bias + gn, acc[im][in]);
    }
}

// -------- decay prep: dfast = mean(sig(fast))^128; msd[h] = mean_dh sig(slow)
__global__ __launch_bounds__(256) void k_decay(const float* __restrict__ fd,
                                               const float* __restrict__ sd,
                                               float* __restrict__ dfast,
                                               float* __restrict__ msd) {
  const int t = threadIdx.x;
  __shared__ float sf[256], ss[256];
  float a = 0.f, b = 0.f;
#pragma unroll
  for (int i = 0; i < 4; ++i) {
    a += sigmoidf_(fd[t * 4 + i]);
    b += sigmoidf_(sd[t * 4 + i]);
  }
  sf[t] = a; ss[t] = b;
  __syncthreads();
  if (t < 16) {
    float s = 0.f;
    for (int j = 0; j < 16; ++j) s += ss[t * 16 + j];
    msd[t] = s / 64.f;
  }
  if (t == 0) {
    float s = 0.f;
    for (int j = 0; j < 256; ++j) s += sf[j];
    dfast[0] = __powf(s / 1024.f, 128.f);
  }
}

// --------- gate / pack: vg = v*sig(g); sg; head-major bf16 outputs ---------
// qh: [B,H,L,DH] row-major (A operand). kh/ksh: per-chunk fragment-major
// (B operand). vT/vsT: [B,H,DH,L] (transposed A operand for V^T @ K).
__global__ __launch_bounds__(256) void k_pack(const float* __restrict__ q,
                                              const float* __restrict__ k,
                                              const float* __restrict__ v,
                                              const float* __restrict__ g,
                                              const float* __restrict__ skw,
                                              const float* __restrict__ skb,
                                              const float* __restrict__ svw,
                                              const float* __restrict__ svb,
                                              __bf16* __restrict__ qh,
                                              __bf16* __restrict__ kh,
                                              __bf16* __restrict__ ksh,
                                              __bf16* __restrict__ vT,
                                              __bf16* __restrict__ vsT,
                                              float* __restrict__ sg_out) {
  const int m = blockIdx.x;            // b*L + l
  const int b = m >> 12;
  const int l = m & (Lseq - 1);
  const int t = threadIdx.x;
  __shared__ float pk[256], pv[256], shs[16];

  float qv[4], kv[4], vv[4];
  float skp = 0.f, svp = 0.f;
  const size_t base = (size_t)m * Dmod + t * 4;
#pragma unroll
  for (int i = 0; i < 4; ++i) {
    const int d  = t * 4 + i;
    const int dh = d & (DH - 1);
    const float kk = k[base + i];
    const float vg = v[base + i] * sigmoidf_(g[base + i]);
    qv[i] = q[base + i]; kv[i] = kk; vv[i] = vg;
    skp += kk * skw[dh];
    svp += vg * svw[dh];
  }
  pk[t] = skp; pv[t] = svp;
  __syncthreads();
  if (t < 16) { // one thread per head
    float a = 0.f, c = 0.f;
    for (int j = 0; j < 16; ++j) { a += pk[t * 16 + j]; c += pv[t * 16 + j]; }
    a += skb[0]; c += svb[0];
    const float s = sigmoidf_(a * c * 10.f);
    shs[t] = s;
    sg_out[((size_t)(b * Hh + t)) * Lseq + l] = s;
  }
  __syncthreads();

  const int h = t >> 4;                 // head of this thread's 4 channels
  const float s = shs[h];
  const size_t bh = (size_t)(b * Hh + h);
  const size_t hb = (bh * Lseq + l) * DH;
  // fragment-major offsets for kh/ksh within this (bh, chunk)
  const int c  = l >> 7;
  const int lc = l & (CHK - 1);
  const int ks = lc >> 5;
  const int j  = lc & 15;
  const int lhalf = ((lc >> 4) & 1) << 4;
  const size_t cbase = ((size_t)bh * Lseq + (size_t)c * CHK) * DH;
#pragma unroll
  for (int i = 0; i < 4; ++i) {
    const int d  = t * 4 + i;
    const int dh = d & (DH - 1);
    qh[hb + dh] = (__bf16)qv[i];
    const size_t ko =
        cbase + (size_t)(((ks * 4 + (dh >> 4)) * 32 + (dh & 15) + lhalf) * 16 + j);
    kh[ko]  = (__bf16)kv[i];
    ksh[ko] = (__bf16)(kv[i] * s);
    vT[(bh * DH + dh) * Lseq + l]  = (__bf16)vv[i];
    vsT[(bh * DH + dh) * Lseq + l] = (__bf16)(vv[i] * s);
  }
}

// -------- per-chunk slow decay: ((1/H) sum_h msd[h]*(1-0.05*mean sg))^128 ---
__global__ __launch_bounds__(256) void k_reduce_sg(const float* __restrict__ sg,
                                                   const float* __restrict__ msd,
                                                   float* __restrict__ dslow) {
  const int c = blockIdx.x;
  const int t = threadIdx.x;
  const int h = t & 15;
  const int j = t >> 4;
  float s = 0.f;
#pragma unroll 4
  for (int it = 0; it < 32; ++it) {
    const int p = j + 16 * it;          // 0..511 over (b,l)
    const int b = p >> 7;
    const int l = p & (CHK - 1);
    s += sg[((size_t)(b * Hh + h)) * Lseq + c * CHK + l];
  }
  __shared__ float hs[256];
  __shared__ float hp[16];
  hs[t] = s;
  __syncthreads();
  if (t < 16) {
    float tot = 0.f;
    for (int jj = 0; jj < 16; ++jj) tot += hs[t + 16 * jj];
    hp[t] = msd[t] * (1.f - 0.05f * (tot / 512.f));
  }
  __syncthreads();
  if (t == 0) {
    float m = 0.f;
    for (int hh = 0; hh < 16; ++hh) m += hp[hh];
    dslow[c] = __powf(m / 16.f, 128.f);
  }
}

// -------------------- dual-path chunked scan, all WMMA ---------------------
// grid.x = B*H, grid.y = path (0 fast, 1 slow). 8 waves keep the 64x64 state
// as 16 distributed f32 accumulators; S round-trips LDS as fragment-major
// bf16 (b128 stores/loads) for the Q@S product.
__global__ __launch_bounds__(256) void k_attn(const __bf16* __restrict__ qh,
                                              const __bf16* __restrict__ kh,
                                              const __bf16* __restrict__ ksh,
                                              const __bf16* __restrict__ vT,
                                              const __bf16* __restrict__ vsT,
                                              const float* __restrict__ dfast,
                                              const float* __restrict__ dslow,
                                              float* __restrict__ ofast,
                                              float* __restrict__ oslow) {
  const int bh   = blockIdx.x;
  const int path = blockIdx.y;
  const __bf16* Kp = path ? ksh : kh;
  const __bf16* Vp = path ? vsT : vT;
  float* O = path ? oslow : ofast;
  const float dconst = dfast[0];
  const int t = threadIdx.x;
  const int w = t >> 5;

  __shared__ __bf16 sQ[CHK * DH];   // 16 KB, [l][d] row-major (A operand)
  __shared__ __bf16 sK[CHK * DH];   // 16 KB, fragment-major 16 tiles (B op)
  __shared__ __bf16 sV[DH * CHK];   // 16 KB, [d][l] (A operand, v transposed)
  __shared__ __bf16 sS[DH * DH];    //  8 KB, fragment-major 8 tiles (B op)

  v8f st[2];
  st[0] = (v8f){0.f, 0.f, 0.f, 0.f, 0.f, 0.f, 0.f, 0.f};
  st[1] = st[0];
  const int tau[2] = {w * 2, w * 2 + 1};

  for (int c = 0; c < NC; ++c) {
    // stage chunk: q and fragment-major k are contiguous 16 KB blocks
    const v4u* gq = (const v4u*)(qh + ((size_t)bh * Lseq + c * CHK) * DH);
    const v4u* gk = (const v4u*)(Kp + ((size_t)bh * Lseq + c * CHK) * DH);
#pragma unroll
    for (int i = t; i < 1024; i += 256) {
      ((v4u*)sQ)[i] = gq[i];
      ((v4u*)sK)[i] = gk[i];
    }
    // vT rows strided by L
#pragma unroll
    for (int i = t; i < 512; i += 256) {
      const int row = i >> 3, c8 = i & 7;
      *(v4u*)(sV + i * 16) =
          *(const v4u*)(Vp + ((size_t)bh * DH + row) * Lseq + c * CHK + c8 * 16);
    }
    __syncthreads();

    const float decay = path ? dslow[c] : dconst;

    // state update: S = S*decay + V^T @ K   (contraction over l = 128)
#pragma unroll
    for (int ti = 0; ti < 2; ++ti) {
      const int dm = tau[ti] >> 2, en = tau[ti] & 3;
      v8f a;
#pragma unroll
      for (int r = 0; r < 8; ++r) a[r] = st[ti][r] * decay;
#pragma unroll
      for (int ks = 0; ks < 4; ++ks) {
        const v16bf af = load_frag_a(&sV[(dm * 16) * CHK + ks * 32], CHK);
        const v16bf bf = load_frag_b_sw(&sK[(ks * 4 + en) * 512]);
        a = wmma_bf16(af, bf, a);
      }
      st[ti] = a;
      store_frag_sS(sS, dm, en, a);
    }
    __syncthreads();

    // o_c = Q @ S : wave w owns rows [w*16, w*16+16)
    const v16bf a0 = load_frag_a(&sQ[(w * 16) * DH + 0], DH);
    const v16bf a1 = load_frag_a(&sQ[(w * 16) * DH + 32], DH);
#pragma unroll
    for (int en = 0; en < 4; ++en) {
      v8f oc = (v8f){0.f, 0.f, 0.f, 0.f, 0.f, 0.f, 0.f, 0.f};
      const v16bf b0 = load_frag_b_sw(&sS[(0 * 4 + en) * 512]);
      const v16bf b1 = load_frag_b_sw(&sS[(1 * 4 + en) * 512]);
      oc = wmma_bf16(a0, b0, oc);
      oc = wmma_bf16(a1, b1, oc);
      store_frag_c(O + ((size_t)bh * Lseq + c * CHK + w * 16) * DH + en * 16,
                   DH, oc);
    }
    __syncthreads();
  }
}

// ------- combine paths with alpha = 0.5+0.3*sg, heads -> [B,L,D] bf16 ------
__global__ __launch_bounds__(256) void k_combine(const float* __restrict__ of,
                                                 const float* __restrict__ os,
                                                 const float* __restrict__ sg,
                                                 __bf16* __restrict__ aO) {
  const size_t idx = (size_t)blockIdx.x * 256 + threadIdx.x; // [B,H,L,DH]
  const int dh = idx & (DH - 1);
  const size_t r = idx >> 6;
  const int l = r & (Lseq - 1);
  const size_t bhi = r >> 12;
  const int h = bhi & (Hh - 1);
  const int b = (int)(bhi >> 4);
  const float s = sg[bhi * Lseq + l];
  const float alpha = 0.5f + 0.3f * s;
  const float val = alpha * of[idx] + (1.f - alpha) * os[idx];
  aO[((size_t)b * Lseq + l) * Dmod + h * DH + dh] = (__bf16)val;
}

// ------------------- residual add + LayerNorm over D -----------------------
__global__ __launch_bounds__(256) void k_layernorm(const float* __restrict__ op,
                                                   const float* __restrict__ x,
                                                   const float* __restrict__ lnw,
                                                   const float* __restrict__ lnb,
                                                   float* __restrict__ out) {
  const int m = blockIdx.x;
  const int t = threadIdx.x;
  __shared__ float s1[256], s2[256];
  float y[4];
  float sum = 0.f, sq = 0.f;
  const size_t base = (size_t)m * Dmod + t * 4;
#pragma unroll
  for (int i = 0; i < 4; ++i) {
    y[i] = op[base + i] + x[base + i];
    sum += y[i]; sq += y[i] * y[i];
  }
  s1[t] = sum; s2[t] = sq;
  __syncthreads();
  for (int off = 128; off > 0; off >>= 1) {
    if (t < off) { s1[t] += s1[t + off]; s2[t] += s2[t + off]; }
    __syncthreads();
  }
  const float mu  = s1[0] * (1.f / Dmod);
  const float var = s2[0] * (1.f / Dmod) - mu * mu;
  const float inv = rsqrtf(var + 1e-5f);
#pragma unroll
  for (int i = 0; i < 4; ++i)
    out[base + i] = (y[i] - mu) * inv * lnw[t * 4 + i] + lnb[t * 4 + i];
}

// ---------------------------------------------------------------------------
extern "C" void kernel_launch(void* const* d_in, const int* in_sizes, int n_in,
                              void* d_out, int out_size, void* d_ws,
                              size_t ws_size, hipStream_t stream) {
  (void)in_sizes; (void)n_in; (void)out_size; (void)ws_size;
  const float* x      = (const float*)d_in[0];
  const float* conv_w = (const float*)d_in[1];
  const float* conv_b = (const float*)d_in[2];
  const float* q_w = (const float*)d_in[3];
  const float* q_b = (const float*)d_in[4];
  const float* k_w = (const float*)d_in[5];
  const float* k_b = (const float*)d_in[6];
  const float* v_w = (const float*)d_in[7];
  const float* v_b = (const float*)d_in[8];
  const float* g_w = (const float*)d_in[9];
  const float* g_b = (const float*)d_in[10];
  const float* o_w = (const float*)d_in[11];
  const float* o_b = (const float*)d_in[12];
  const float* sk_w = (const float*)d_in[13];
  const float* sk_b = (const float*)d_in[14];
  const float* sv_w = (const float*)d_in[15];
  const float* sv_b = (const float*)d_in[16];
  const float* fast_decay = (const float*)d_in[17];
  const float* slow_decay = (const float*)d_in[18];
  const float* ln_w = (const float*)d_in[19];
  const float* ln_b = (const float*)d_in[20];
  float* out = (float*)d_out;

  char* ws = (char*)d_ws;
  const size_t SZ_W   = (size_t)Dmod * Dmod * 2;   // 2 MB bf16 weight
  const size_t SZ_MDb = (size_t)Mrows * Dmod * 2;  // 32 MB bf16 activation
  const size_t SZ_MDf = (size_t)Mrows * Dmod * 4;  // 64 MB fp32 activation

  __bf16* wtQ = (__bf16*)(ws);
  __bf16* wtK = (__bf16*)(ws + SZ_W);
  __bf16* wtV = (__bf16*)(ws + 2 * SZ_W);
  __bf16* wtG = (__bf16*)(ws + 3 * SZ_W);
  __bf16* wtO = (__bf16*)(ws + 4 * SZ_W);
  const size_t OFF_XC = 5 * SZ_W;
  __bf16* xc = (__bf16*)(ws + OFF_XC);             // later reused as aO
  const size_t OFF_Q = OFF_XC + SZ_MDb;
  float* qf = (float*)(ws + OFF_Q);
  float* kf = (float*)(ws + OFF_Q + SZ_MDf);
  float* vf = (float*)(ws + OFF_Q + 2 * SZ_MDf);
  float* gf = (float*)(ws + OFF_Q + 3 * SZ_MDf);
  const size_t OFF_P = OFF_Q + 4 * SZ_MDf;
  __bf16* qh  = (__bf16*)(ws + OFF_P);
  __bf16* kh  = (__bf16*)(ws + OFF_P + SZ_MDb);
  __bf16* ksh = (__bf16*)(ws + OFF_P + 2 * SZ_MDb);
  __bf16* vT  = (__bf16*)(ws + OFF_P + 3 * SZ_MDb);
  __bf16* vsT = (__bf16*)(ws + OFF_P + 4 * SZ_MDb);
  const size_t OFF_SG = OFF_P + 5 * SZ_MDb;
  float* sg    = (float*)(ws + OFF_SG);            // B*H*L fp32 (1 MB)
  float* msd   = (float*)(ws + OFF_SG + (size_t)Bsz * Hh * Lseq * 4);
  float* dfast = msd + 16;
  float* dslow = dfast + 1;                        // 32 floats
  // aliases (fp32 q/k/v/g dead after k_pack)
  float* ofast = qf;
  float* oslow = kf;
  float* oproj = vf;
  __bf16* aO = xc;

  // 1. weights -> bf16 fragment-major
  const int wblocks = (Dmod * Dmod) / 256;
  k_prep_w<<<wblocks, 256, 0, stream>>>(q_w, wtQ);
  k_prep_w<<<wblocks, 256, 0, stream>>>(k_w, wtK);
  k_prep_w<<<wblocks, 256, 0, stream>>>(v_w, wtV);
  k_prep_w<<<wblocks, 256, 0, stream>>>(g_w, wtG);
  k_prep_w<<<wblocks, 256, 0, stream>>>(o_w, wtO);

  // 2. depthwise conv + SiLU -> bf16 A matrix
  k_conv_silu<<<(Mrows * Dmod) / 256, 256, 0, stream>>>(x, conv_w, conv_b, xc);

  // 3. q/k/v/g projections (WMMA GEMM)
  const dim3 ggrid(Dmod / 128, Mrows / 128);
  k_gemm_bf16<<<ggrid, 256, 0, stream>>>(xc, wtQ, q_b, qf, Dmod);
  k_gemm_bf16<<<ggrid, 256, 0, stream>>>(xc, wtK, k_b, kf, Dmod);
  k_gemm_bf16<<<ggrid, 256, 0, stream>>>(xc, wtV, v_b, vf, Dmod);
  k_gemm_bf16<<<ggrid, 256, 0, stream>>>(xc, wtG, g_b, gf, Dmod);

  // 4. decay constants; 5. gate/pack; 6. per-chunk slow decay
  k_decay<<<1, 256, 0, stream>>>(fast_decay, slow_decay, dfast, msd);
  k_pack<<<Mrows, 256, 0, stream>>>(qf, kf, vf, gf, sk_w, sk_b, sv_w, sv_b,
                                    qh, kh, ksh, vT, vsT, sg);
  k_reduce_sg<<<NC, 256, 0, stream>>>(sg, msd, dslow);

  // 7. dual-path chunked linear-attention scan (WMMA)
  k_attn<<<dim3(Bsz * Hh, 2), 256, 0, stream>>>(qh, kh, ksh, vT, vsT,
                                                dfast, dslow, ofast, oslow);

  // 8. blend + pack to [B,L,D] bf16; 9. output projection; 10. LayerNorm
  k_combine<<<(Mrows * Dmod) / 256, 256, 0, stream>>>(ofast, oslow, sg, aO);
  k_gemm_bf16<<<ggrid, 256, 0, stream>>>(aO, wtO, o_b, oproj, Dmod);
  k_layernorm<<<Mrows, 256, 0, stream>>>(oproj, x, ln_w, ln_b, out);
}
